// Multihead_Attention_773094113483
// MI455X (gfx1250) — matrix operand order repaired and checked by reading, not compiled, not run on hardware
//
#include <hip/hip_runtime.h>
#include <hip/hip_bf16.h>

// ---------------------------------------------------------------------------
// CDNA5 (gfx1250) bf16 WMMA co-attention pipeline.
//   N=64, TC=512, TQ=64, C=HID=512, H=8, HD=HDV=64
// ---------------------------------------------------------------------------

#define DN   64
#define DTC  512
#define DTQ  64
#define DC   512
#define DHID 512
#define DH   8
#define DHD  64
#define FNEG (-1e30f)

typedef __attribute__((ext_vector_type(16))) __bf16 v16bf;
typedef __attribute__((ext_vector_type(8)))  float  v8f;

union BF16x16 {
    v16bf v;
    __bf16 e[16];
    unsigned int u[8];
};

// Native conversion -> backend emits hardware (packed) f32->bf16 converts.
__device__ __forceinline__ __bf16 f2bf(float f) { return (__bf16)f; }

__device__ __forceinline__ v8f wmma_bf16(v16bf a, v16bf b, v8f c) {
    // (neg_a, A, neg_b, B, c_mod, C, reuse_a, reuse_b)
    return __builtin_amdgcn_wmma_f32_16x16x32_bf16(
        false, a, false, b, (short)0, c, false, false);
}

// ---------------------------------------------------------------------------
// Kernel A: Out[M,Nout] = relu(A[M,K] @ W[Nout,K]^T + bias), bf16 output.
// Block = 256 threads (8 waves), 64x64 output tile, K in steps of 32.
// Raw float4 loads first (all in flight), then packed converts, then WMMA.
// ---------------------------------------------------------------------------
__global__ void __launch_bounds__(256)
proj_gemm_relu_bf16(const float* __restrict__ A,     // [M, K]
                    const float* __restrict__ W,     // [Nout, K]
                    const float* __restrict__ bias,  // [Nout]
                    __bf16* __restrict__ Out,        // [M, Nout]
                    int M, int K, int Nout) {
    const int lane = threadIdx.x & 31;
    const int wave = threadIdx.x >> 5;
    const int ln   = lane & 15;
    const int hi   = lane >> 4;                       // 0 | 1 (lane half)

    const int mBase = blockIdx.x * 64 + (wave & 3) * 16;
    const int nBlk  = blockIdx.y * 64;
    const int ni0   = wave >> 2;                      // 0 | 1

    const int row   = mBase + ln;                     // A-fragment row (per lane)
    const int ncol0 = nBlk + ni0 * 16 + ln;           // B/D column, tile 0
    const int ncol1 = nBlk + (ni0 + 2) * 16 + ln;     // B/D column, tile 1

    v8f acc0 = {}; v8f acc1 = {};

    for (int kb = 0; kb < K; kb += 32) {
        const float* Arow = A + (size_t)row * K + kb;
        const float* W0   = W + (size_t)ncol0 * K + kb + hi * 16;
        const float* W1   = W + (size_t)ncol1 * K + kb + hi * 16;

        if (kb + 32 < K) {                            // gfx1250 global_prefetch_b8
            __builtin_prefetch(Arow + 32, 0, 0);
            __builtin_prefetch(W0 + 32, 0, 0);
            __builtin_prefetch(W1 + 32, 0, 0);
        }

        // ---- raw loads (12 x b128, overlapped) ----
        float4 ar[4], b0r[4], b1r[4];
        ar[0] = *reinterpret_cast<const float4*>(Arow + hi * 16);
        ar[1] = *reinterpret_cast<const float4*>(Arow + hi * 16 + 4);
        ar[2] = *reinterpret_cast<const float4*>(Arow + hi * 16 + 8);
        ar[3] = *reinterpret_cast<const float4*>(Arow + hi * 16 + 12);
#pragma unroll
        for (int j = 0; j < 4; ++j) {
            b0r[j] = *reinterpret_cast<const float4*>(W0 + 4 * j);
            b1r[j] = *reinterpret_cast<const float4*>(W1 + 4 * j);
        }

        // ---- packed f32->bf16 converts into WMMA fragment layout ----
        // A 16x32: e[i]    = K hi*16+i                     (flat order)
        // B 32x16: e[i]    = K hi*16+i                     (flat order)
        BF16x16 a, b0, b1;
        const float* af = reinterpret_cast<const float*>(ar);
        const float* f0 = reinterpret_cast<const float*>(b0r);
        const float* f1 = reinterpret_cast<const float*>(b1r);
#pragma unroll
        for (int i = 0; i < 16; ++i) {
            a.e[i]  = f2bf(af[i]);
            b0.e[i] = f2bf(f0[i]);
            b1.e[i] = f2bf(f1[i]);
        }
        acc0 = wmma_bf16(a.v, b0.v, acc0);
        acc1 = wmma_bf16(a.v, b1.v, acc1);
    }

    const float bs0 = bias[ncol0];
    const float bs1 = bias[ncol1];
#pragma unroll
    for (int v = 0; v < 8; ++v) {
        const int m = mBase + v + hi * 8;             // C/D layout: M = v + hi*8
        float x0 = acc0[v] + bs0; x0 = x0 > 0.f ? x0 : 0.f;
        float x1 = acc1[v] + bs1; x1 = x1 > 0.f ? x1 : 0.f;
        Out[(size_t)m * Nout + ncol0] = f2bf(x0);
        Out[(size_t)m * Nout + ncol1] = f2bf(x1);
    }
}

// ---------------------------------------------------------------------------
// Kernel B: one workgroup per (head h, batch n). Whole (h,n) problem in LDS.
//   S = mask( Q@K^T / 8 ) : [512 x 64] f32 in LDS
//   cw = softmax_rows(S), qw = softmax_cols(S)    (recomputed from S + stats)
//   question_att = qw^T @ Q        -> out + LDS (bf16 copy)
//   context_att  = cw @ V          -> out
//   context_coatt= cw @ question_att -> out
// LDS budget: 236.5 KB dynamic (WGP has 320 KB).
// ---------------------------------------------------------------------------
__global__ void __launch_bounds__(256)
coattn_kernel(const __bf16* __restrict__ Qbf,   // [N*TC, HID] bf16
              const __bf16* __restrict__ Kbf,   // [N*TQ, HID] bf16
              const __bf16* __restrict__ Vbf,   // [N*TQ, C]   bf16
              const int*    __restrict__ cmask, // [N*TC]
              const int*    __restrict__ qmask, // [N*TQ]
              float* __restrict__ outCtx,       // [N, TC, C]
              float* __restrict__ outQatt,      // [N, TQ, HID]
              float* __restrict__ outCoatt) {   // [N, TC, HID]
    extern __shared__ __align__(16) char smem[];
    float*  sS    = reinterpret_cast<float*>(smem);                 // 512*64 f32  (128 KB)
    float*  sQA   = reinterpret_cast<float*>(smem + 131072);        // 64*64 f32   (16 KB)
    float*  srmax = reinterpret_cast<float*>(smem + 147456);        // 512 f32
    float*  srinv = srmax + DTC;                                    // 512 f32
    float*  scmax = srinv + DTC;                                    // 64 f32
    float*  scinv = scmax + DTQ;                                    // 64 f32
    __bf16* sQ    = reinterpret_cast<__bf16*>(scinv + DTQ);         // 512*64 bf16 (64 KB)
    __bf16* sK    = sQ + DTC * DHD;                                 // 64*64 bf16
    __bf16* sV    = sK + DTQ * DHD;                                 // 64*64 bf16
    __bf16* sQAb  = sV + DTQ * DHD;                                 // 64*64 bf16

    const int h    = blockIdx.x & 7;
    const int n    = blockIdx.x >> 3;
    const int tid  = threadIdx.x;
    const int lane = tid & 31;
    const int wave = tid >> 5;
    const int ln   = lane & 15;
    const int hi   = lane >> 4;

    // ---- Phase 1: stage Q, K, V head slices into LDS (16B vector copies) ----
    const __bf16* gQ = Qbf + (size_t)n * DTC * DHID + h * DHD;
    const __bf16* gK = Kbf + (size_t)n * DTQ * DHID + h * DHD;
    const __bf16* gV = Vbf + (size_t)n * DTQ * DC   + h * DHD;
    for (int i = tid; i < DTC * 8; i += 256) {       // 512 rows x 8 uint4
        const int r = i >> 3, c = (i & 7) * 8;
        *reinterpret_cast<uint4*>(sQ + r * DHD + c) =
            *reinterpret_cast<const uint4*>(gQ + (size_t)r * DHID + c);
    }
    for (int i = tid; i < DTQ * 8; i += 256) {       // 64 rows x 8 uint4, K and V
        const int r = i >> 3, c = (i & 7) * 8;
        *reinterpret_cast<uint4*>(sK + r * DHD + c) =
            *reinterpret_cast<const uint4*>(gK + (size_t)r * DHID + c);
        *reinterpret_cast<uint4*>(sV + r * DHD + c) =
            *reinterpret_cast<const uint4*>(gV + (size_t)r * DC + c);
    }
    __syncthreads();

    // ---- Phase 2: S = mask( (Q @ K^T) * 1/sqrt(HD) ) ----
    const float scale = 0.125f;                      // 1/sqrt(64)
    for (int tile = wave; tile < 32 * 4; tile += 8) {
        const int mt = tile >> 2, nt = tile & 3;
        const int arow = mt * 16 + ln;
        const int kcol = nt * 16 + ln;
        v8f acc = {};
        for (int kb = 0; kb < DHD; kb += 32) {
            BF16x16 a, b;
#pragma unroll
            for (int v = 0; v < 8; ++v) {
                const int ka = kb + hi * 16 + 2 * v;
                a.u[v] = *reinterpret_cast<const unsigned*>(sQ + arow * DHD + ka);
                const int kx = kb + hi * 16 + 2 * v;          // B[k,n] = K[kcol][k]
                b.u[v] = *reinterpret_cast<const unsigned*>(sK + kcol * DHD + kx);
            }
            acc = wmma_bf16(a.v, b.v, acc);
        }
        const int qm = qmask[n * DTQ + kcol];
#pragma unroll
        for (int v = 0; v < 8; ++v) {
            const int t = mt * 16 + v + hi * 8;
            const int cm = cmask[n * DTC + t];
            sS[t * DTQ + kcol] = (cm & qm) ? acc[v] * scale : FNEG;
        }
    }
    __syncthreads();

    // ---- Phase 3: softmax statistics (rows for cw, columns for qw) ----
    for (int r = tid; r < DTC; r += 256) {
        float mx = -3.4e38f;
        for (int k = 0; k < DTQ; ++k) mx = fmaxf(mx, sS[r * DTQ + k]);
        float sum = 0.f;
        for (int k = 0; k < DTQ; ++k) sum += __expf(sS[r * DTQ + k] - mx);
        srmax[r] = mx;
        srinv[r] = 1.0f / sum;
    }
    if (tid < DTQ) {
        const int k = tid;
        float mx = -3.4e38f;
        for (int r = 0; r < DTC; ++r) mx = fmaxf(mx, sS[r * DTQ + k]);
        float sum = 0.f;
        for (int r = 0; r < DTC; ++r) sum += __expf(sS[r * DTQ + k] - mx);
        scmax[k] = mx;
        scinv[k] = 1.0f / sum;
    }
    __syncthreads();

    // ---- Phase 4: question_att[k,d] = sum_q qw[q,k] * Q[q,d]  (qw^T @ Q) ----
    for (int tile = wave; tile < 16; tile += 8) {
        const int kt = tile >> 2, dt = tile & 3;
        const int kcol = kt * 16 + ln;               // A-fragment row index (over k)
        const int dcol = dt * 16 + ln;               // B/D column (over d)
        const float cmx = scmax[kcol], cin = scinv[kcol];
        v8f acc = {};
        for (int qb = 0; qb < DTC; qb += 32) {
            BF16x16 a, b;
#pragma unroll
            for (int v = 0; v < 8; ++v) {
                const int q0 = qb + hi * 16 + 2 * v;
                a.e[2 * v]     = f2bf(__expf(sS[q0 * DTQ + kcol] - cmx) * cin);
                a.e[2 * v + 1] = f2bf(__expf(sS[(q0 + 1) * DTQ + kcol] - cmx) * cin);
                const int q1 = qb + hi * 16 + 2 * v; // B[q,d] = Q[q][dcol]
                b.e[2 * v]     = sQ[q1 * DHD + dcol];
                b.e[2 * v + 1] = sQ[(q1 + 1) * DHD + dcol];
            }
            acc = wmma_bf16(a.v, b.v, acc);
        }
#pragma unroll
        for (int v = 0; v < 8; ++v) {
            const int kk = kt * 16 + v + hi * 8;
            sQA[kk * DHD + dcol]  = acc[v];
            sQAb[kk * DHD + dcol] = f2bf(acc[v]);
            outQatt[((size_t)n * DTQ + kk) * DHID + h * DHD + dcol] = acc[v];
        }
    }
    __syncthreads();

    // ---- Phase 5: context_att = cw @ V, context_coatt = cw @ question_att ----
    for (int tile = wave; tile < 32 * 4; tile += 8) {
        const int mt = tile >> 2, dt = tile & 3;
        const int trow = mt * 16 + ln;
        const int dcol = dt * 16 + ln;
        const float rmx = srmax[trow], rin = srinv[trow];
        v8f acc1 = {}, acc2 = {};
        for (int kb = 0; kb < DTQ; kb += 32) {
            BF16x16 a, bv, bq;
#pragma unroll
            for (int v = 0; v < 8; ++v) {
                const int ka = kb + hi * 16 + 2 * v;
                const float2 s2 = *reinterpret_cast<const float2*>(sS + trow * DTQ + ka);
                a.e[2 * v]     = f2bf(__expf(s2.x - rmx) * rin);
                a.e[2 * v + 1] = f2bf(__expf(s2.y - rmx) * rin);
                const int kx = kb + hi * 16 + 2 * v;
                bv.e[2 * v]     = sV[kx * DHD + dcol];
                bv.e[2 * v + 1] = sV[(kx + 1) * DHD + dcol];
                bq.e[2 * v]     = sQAb[kx * DHD + dcol];
                bq.e[2 * v + 1] = sQAb[(kx + 1) * DHD + dcol];
            }
            acc1 = wmma_bf16(a.v, bv.v, acc1);       // shared A fragment
            acc2 = wmma_bf16(a.v, bq.v, acc2);
        }
#pragma unroll
        for (int v = 0; v < 8; ++v) {
            const int t = mt * 16 + v + hi * 8;
            const size_t o = ((size_t)n * DTC + t) * DC + h * DHD + dcol;
            outCtx[o]   = acc1[v];
            outCoatt[o] = acc2[v];
        }
    }
}

// ---------------------------------------------------------------------------
extern "C" void kernel_launch(void* const* d_in, const int* in_sizes, int n_in,
                              void* d_out, int out_size, void* d_ws, size_t ws_size,
                              hipStream_t stream) {
    const float* Context  = (const float*)d_in[0];   // [64,512,512]
    const float* Question = (const float*)d_in[1];   // [64,64,512]
    const int*   cmask    = (const int*)d_in[2];     // [64,512,1]
    const int*   qmask    = (const int*)d_in[3];     // [64,1,64]
    const float* WQ       = (const float*)d_in[4];
    const float* bQ       = (const float*)d_in[5];
    const float* WK       = (const float*)d_in[6];
    const float* bK       = (const float*)d_in[7];
    const float* WV       = (const float*)d_in[8];
    const float* bV       = (const float*)d_in[9];

    __bf16* Qbf = (__bf16*)d_ws;                          // 64*512*512 bf16
    __bf16* Kbf = Qbf + (size_t)DN * DTC * DHID;          // 64*64*512 bf16
    __bf16* Vbf = Kbf + (size_t)DN * DTQ * DHID;          // 64*64*512 bf16

    const dim3 blk(256);
    // Projections: relu(X @ W^T + b) -> bf16
    proj_gemm_relu_bf16<<<dim3(DN * DTC / 64, DHID / 64), blk, 0, stream>>>(
        Context, WQ, bQ, Qbf, DN * DTC, DC, DHID);
    proj_gemm_relu_bf16<<<dim3(DN * DTQ / 64, DHID / 64), blk, 0, stream>>>(
        Question, WK, bK, Kbf, DN * DTQ, DC, DHID);
    proj_gemm_relu_bf16<<<dim3(DN * DTQ / 64, DC / 64), blk, 0, stream>>>(
        Question, WV, bV, Vbf, DN * DTQ, DC, DC);

    float* outCtx   = (float*)d_out;
    float* outQatt  = outCtx  + (size_t)DN * DTC * DC;
    float* outCoatt = outQatt + (size_t)DN * DTQ * DHID;

    // LDS: 128K (S) + 16K (QA) + 4.5K (stats) + 64K (Q) + 8K*3 (K,V,QAb) = 242176 B
    const size_t smemB = 242176;
    coattn_kernel<<<dim3(DN * DH), blk, smemB, stream>>>(
        Qbf, Kbf, Vbf, cmask, qmask, outCtx, outQatt, outCoatt);
}
